// KnowledgeGraphEncoder_15857019257425
// MI455X (gfx1250) — matrix operand order repaired and checked by reading, not compiled, not run on hardware
//
#include <hip/hip_runtime.h>
#include <stdint.h>

#define NNODES 4096
#define INDIM  256
#define HID    64
#define OUTD   128
#define NHEADS 4

typedef __attribute__((ext_vector_type(16))) _Float16 v16h;
typedef __attribute__((ext_vector_type(8)))  float    v8f;

union V16 { v16h v; _Float16 h[16]; uint4 q[2]; };

static __device__ __forceinline__ v8f wmma16(v16h a, v16h b, v8f c) {
  return __builtin_amdgcn_wmma_f32_16x16x32_f16(false, a, false, b, (short)0, c,
                                                false, false);
}

// B fragment (32x16): lane (n=lane&15, kh=lane>>4) holds K values 8kh..8kh+7 and
// 16+8kh..16+8kh+7 of column n -> two 16B loads from a [col][node] (transposed) f16 matrix.
static __device__ __forceinline__ v16h bload(const _Float16* p) {
  V16 b;
  b.q[0] = *(const uint4*)p;
  b.q[1] = *(const uint4*)(p + 16);
  return b.v;
}

static __device__ __forceinline__ float pterm(int msk, float dv, float si, float mrow) {
  if (msk <= 0) return 0.f;
  float e = si + dv;
  e = e > 0.f ? e : 0.2f * e;               // leaky_relu(0.2)
  return __expf(e - mrow);
}

// ---- W transpose + f16 convert: WT[head][col][k] = W[head][k][col] ---------
__global__ void gat_wprep(const float* __restrict__ W, _Float16* __restrict__ WT,
                          int K, int ncols, int nheads) {
  const int g = blockIdx.x * 256 + threadIdx.x;
  if (g >= nheads * ncols * K) return;
  const int k = g % K, rest = g / K;
  const int c = rest % ncols, h = rest / ncols;
  WT[g] = (_Float16)W[((size_t)h * K + k) * ncols + c];
}

// ---- Projection, LDS-free. Block = 16 rows x 64 cols; wave w = 16 cols. ----
// Outputs: row-major f32 [head*rows+row][NCOLS] and transposed f16
// [head*NCOLS+col][rows] (B-operand layout for the attention kernels).
template <int K, int NCOLS>
__global__ __launch_bounds__(128) void gat_proj(
    const float* __restrict__ A, const _Float16* __restrict__ WT,
    float* __restrict__ oF32, _Float16* __restrict__ oT16, int rows) {
  const int tid = threadIdx.x, lane = tid & 31, w = tid >> 5;
  const int row0 = blockIdx.x * 16;
  constexpr int CG = NCOLS / 64;
  const int head  = blockIdx.y / CG;
  const int cbase = (blockIdx.y % CG) * 64;
  const int m = lane & 15, kh = lane >> 4;
  const int col = cbase + w * 16 + m;       // n == lane&15
  const float*    arow  = A + (size_t)(row0 + m) * K;
  const _Float16* wtrow = WT + ((size_t)head * NCOLS + col) * K;

  v8f acc = {};
#pragma unroll
  for (int k0 = 0; k0 < K; k0 += 32) {
    const float4* xa = (const float4*)(arow + k0 + kh * 8);
    const float4* xb = (const float4*)(arow + k0 + 16 + kh * 8);
    const float4 x0 = xa[0], x1 = xa[1], x2 = xb[0], x3 = xb[1];
    V16 Af;
    Af.h[0]  = (_Float16)x0.x; Af.h[1]  = (_Float16)x0.y;
    Af.h[2]  = (_Float16)x0.z; Af.h[3]  = (_Float16)x0.w;
    Af.h[4]  = (_Float16)x1.x; Af.h[5]  = (_Float16)x1.y;
    Af.h[6]  = (_Float16)x1.z; Af.h[7]  = (_Float16)x1.w;
    Af.h[8]  = (_Float16)x2.x; Af.h[9]  = (_Float16)x2.y;
    Af.h[10] = (_Float16)x2.z; Af.h[11] = (_Float16)x2.w;
    Af.h[12] = (_Float16)x3.x; Af.h[13] = (_Float16)x3.y;
    Af.h[14] = (_Float16)x3.z; Af.h[15] = (_Float16)x3.w;
    acc = wmma16(Af.v, bload(wtrow + k0 + kh * 8), acc);
  }
  // C fragment: vgpr r -> row (r + 8*kh), col fixed per lane.
#pragma unroll
  for (int r = 0; r < 8; ++r)
    oF32[(size_t)(head * rows + row0 + r + 8 * kh) * NCOLS + col] = acc[r];
  union { uint4 q; _Float16 h[8]; } st;
#pragma unroll
  for (int r = 0; r < 8; ++r) st.h[r] = (_Float16)acc[r];
  *(uint4*)(oT16 + ((size_t)head * NCOLS + col) * rows + row0 + 8 * kh) = st.q;
}

// ---- src/dst attention coefficients ----------------------------------------
__global__ void gat_srcdst(const float* __restrict__ hmat,
                           const float* __restrict__ avec,
                           float* __restrict__ src, float* __restrict__ dst,
                           int n, int F, int nheads) {
  const int g = blockIdx.x * blockDim.x + threadIdx.x;
  if (g >= n * nheads) return;
  const int head = g / n;
  const float* hr = hmat + (size_t)g * F;
  const float* a  = avec + (size_t)head * 2 * F;
  float s = 0.f, d = 0.f;
  for (int f = 0; f < F; ++f) { s += hr[f] * a[f]; d += hr[f] * a[F + f]; }
  src[g] = s; dst[g] = d;
}

// ---- per-head global max of dst (softmax stabilizer bound) -----------------
__global__ __launch_bounds__(256) void gat_colmax(const float* __restrict__ dst,
                                                  float* __restrict__ out, int n) {
  __shared__ float red[256];
  const float* d = dst + (size_t)blockIdx.x * n;
  float m = -3.4e38f;
  for (int i = threadIdx.x; i < n; i += 256) m = fmaxf(m, d[i]);
  red[threadIdx.x] = m;
  __syncthreads();
  for (int s = 128; s > 0; s >>= 1) {
    if (threadIdx.x < s) red[threadIdx.x] = fmaxf(red[threadIdx.x], red[threadIdx.x + s]);
    __syncthreads();
  }
  if (threadIdx.x == 0) out[blockIdx.x] = red[0];
}

#define PT(i, M, D) \
  do { float p_ = pterm((M), (D), si, mrow); Af.h[i] = (_Float16)p_; part += p_; } while (0)

// ---- Layer-1 fused attention: softmax(mask(e)) @ h, 4 heads, +ELU ----------
// Block = 16 rows; wave w = head. Register-only: A frag (P tile) is computed
// directly in WMMA layout; B frags load from transposed h. No LDS, no barriers.
__global__ __launch_bounds__(128) void gat_attn1(
    const int* __restrict__ adj, const float* __restrict__ src,
    const float* __restrict__ dst, const float* __restrict__ dmaxv,
    const _Float16* __restrict__ hT, float* __restrict__ hcat) {
  const int tid = threadIdx.x, lane = tid & 31, h = tid >> 5;
  const int row0 = blockIdx.x * 16;
  const int pi = lane & 15, kh = lane >> 4;
  const float si = src[h * NNODES + row0 + pi];
  float mrow = si + dmaxv[h];
  mrow = mrow > 0.f ? mrow : 0.2f * mrow;
  const int*      arow = adj + (size_t)(row0 + pi) * NNODES;
  const float*    drow = dst + (size_t)h * NNODES;
  const _Float16* hTh  = hT + (size_t)h * HID * NNODES;
  const int n = lane & 15;

  v8f acc0 = {}, acc1 = {}, acc2 = {}, acc3 = {};
  float part = 0.f;
#pragma unroll 2
  for (int j0 = 0; j0 < NNODES; j0 += 32) {
    const int c0 = j0 + kh * 8;   // A frag cols: c0..c0+7 and c0+16..c0+23
    __builtin_prefetch(arow + j0 + 128, 0, 1);
    const int4   m0 = *(const int4*)(arow + c0);
    const int4   m1 = *(const int4*)(arow + c0 + 4);
    const int4   m2 = *(const int4*)(arow + c0 + 16);
    const int4   m3 = *(const int4*)(arow + c0 + 20);
    const float4 d0 = *(const float4*)(drow + c0);
    const float4 d1 = *(const float4*)(drow + c0 + 4);
    const float4 d2 = *(const float4*)(drow + c0 + 16);
    const float4 d3 = *(const float4*)(drow + c0 + 20);
    V16 Af;
    PT(0,  m0.x, d0.x); PT(1,  m0.y, d0.y); PT(2,  m0.z, d0.z); PT(3,  m0.w, d0.w);
    PT(4,  m1.x, d1.x); PT(5,  m1.y, d1.y); PT(6,  m1.z, d1.z); PT(7,  m1.w, d1.w);
    PT(8,  m2.x, d2.x); PT(9,  m2.y, d2.y); PT(10, m2.z, d2.z); PT(11, m2.w, d2.w);
    PT(12, m3.x, d3.x); PT(13, m3.y, d3.y); PT(14, m3.z, d3.z); PT(15, m3.w, d3.w);
    const _Float16* bp = hTh + j0 + kh * 8;
    acc0 = wmma16(Af.v, bload(bp + (size_t)(n)      * NNODES), acc0);
    acc1 = wmma16(Af.v, bload(bp + (size_t)(16 + n) * NNODES), acc1);
    acc2 = wmma16(Af.v, bload(bp + (size_t)(32 + n) * NNODES), acc2);
    acc3 = wmma16(Af.v, bload(bp + (size_t)(48 + n) * NNODES), acc3);
  }
  const float tot = part + __shfl_xor(part, 16);  // full row sum for row pi
#pragma unroll
  for (int r = 0; r < 8; ++r) {
    const int m = r + 8 * kh;
    const float inv = 1.0f / __shfl(tot, m);
    float* orow = hcat + (size_t)(row0 + m) * (NHEADS * HID) + h * HID;
    float v0 = acc0[r] * inv; orow[n]      = v0 > 0.f ? v0 : __expf(v0) - 1.f;
    float v1 = acc1[r] * inv; orow[16 + n] = v1 > 0.f ? v1 : __expf(v1) - 1.f;
    float v2 = acc2[r] * inv; orow[32 + n] = v2 > 0.f ? v2 : __expf(v2) - 1.f;
    float v3 = acc3[r] * inv; orow[48 + n] = v3 > 0.f ? v3 : __expf(v3) - 1.f;
  }
}

// ---- Layer-2 fused attention: single head, 128 cols; wave w = 32 cols ------
__global__ __launch_bounds__(128) void gat_attn2(
    const int* __restrict__ adj, const float* __restrict__ src,
    const float* __restrict__ dst, const float* __restrict__ dmaxv,
    const _Float16* __restrict__ hT, float* __restrict__ out) {
  const int tid = threadIdx.x, lane = tid & 31, w = tid >> 5;
  const int row0 = blockIdx.x * 16;
  const int pi = lane & 15, kh = lane >> 4;
  const float si = src[row0 + pi];
  float mrow = si + dmaxv[0];
  mrow = mrow > 0.f ? mrow : 0.2f * mrow;
  const int* arow = adj + (size_t)(row0 + pi) * NNODES;
  const int n = lane & 15;

  v8f acc0 = {}, acc1 = {};
  float part = 0.f;
#pragma unroll 2
  for (int j0 = 0; j0 < NNODES; j0 += 32) {
    const int c0 = j0 + kh * 8;
    __builtin_prefetch(arow + j0 + 128, 0, 1);
    const int4   m0 = *(const int4*)(arow + c0);
    const int4   m1 = *(const int4*)(arow + c0 + 4);
    const int4   m2 = *(const int4*)(arow + c0 + 16);
    const int4   m3 = *(const int4*)(arow + c0 + 20);
    const float4 d0 = *(const float4*)(dst + c0);
    const float4 d1 = *(const float4*)(dst + c0 + 4);
    const float4 d2 = *(const float4*)(dst + c0 + 16);
    const float4 d3 = *(const float4*)(dst + c0 + 20);
    V16 Af;
    PT(0,  m0.x, d0.x); PT(1,  m0.y, d0.y); PT(2,  m0.z, d0.z); PT(3,  m0.w, d0.w);
    PT(4,  m1.x, d1.x); PT(5,  m1.y, d1.y); PT(6,  m1.z, d1.z); PT(7,  m1.w, d1.w);
    PT(8,  m2.x, d2.x); PT(9,  m2.y, d2.y); PT(10, m2.z, d2.z); PT(11, m2.w, d2.w);
    PT(12, m3.x, d3.x); PT(13, m3.y, d3.y); PT(14, m3.z, d3.z); PT(15, m3.w, d3.w);
    const _Float16* bp = hT + j0 + kh * 8;
    acc0 = wmma16(Af.v, bload(bp + (size_t)(32 * w + n)      * NNODES), acc0);
    acc1 = wmma16(Af.v, bload(bp + (size_t)(32 * w + 16 + n) * NNODES), acc1);
  }
  const float tot = part + __shfl_xor(part, 16);
#pragma unroll
  for (int r = 0; r < 8; ++r) {
    const int m = r + 8 * kh;
    const float inv = 1.0f / __shfl(tot, m);
    float* orow = out + (size_t)(row0 + m) * OUTD + 32 * w;
    orow[n]      = acc0[r] * inv;
    orow[16 + n] = acc1[r] * inv;
  }
}

extern "C" void kernel_launch(void* const* d_in, const int* in_sizes, int n_in,
                              void* d_out, int out_size, void* d_ws, size_t ws_size,
                              hipStream_t stream) {
  (void)in_sizes; (void)n_in; (void)out_size; (void)ws_size;
  const float* X   = (const float*)d_in[0];
  const int*   adj = (const int*)d_in[1];
  const float* W1  = (const float*)d_in[2];
  const float* a1  = (const float*)d_in[3];
  const float* W2  = (const float*)d_in[4];
  const float* a2  = (const float*)d_in[5];
  float* out = (float*)d_out;

  size_t off = 0;
  auto carve = [&](size_t bytes) -> void* {
    void* p = (char*)d_ws + off;
    off += (bytes + 255) & ~(size_t)255;
    return p;
  };
  _Float16* WT1   = (_Float16*)carve((size_t)NHEADS * INDIM * HID * 2);
  _Float16* WT2   = (_Float16*)carve((size_t)(NHEADS * HID) * OUTD * 2);
  float*    h1f32 = (float*)carve((size_t)NHEADS * NNODES * HID * 4);
  _Float16* h1T   = (_Float16*)carve((size_t)NHEADS * HID * NNODES * 2);
  float*    src1  = (float*)carve((size_t)NHEADS * NNODES * 4);
  float*    dst1  = (float*)carve((size_t)NHEADS * NNODES * 4);
  float*    dmax1 = (float*)carve(NHEADS * 4);
  float*    hcat  = (float*)carve((size_t)NNODES * NHEADS * HID * 4);
  float*    h2f32 = (float*)carve((size_t)NNODES * OUTD * 4);
  _Float16* h2T   = (_Float16*)carve((size_t)OUTD * NNODES * 2);
  float*    src2  = (float*)carve(NNODES * 4);
  float*    dst2  = (float*)carve(NNODES * 4);
  float*    dmax2 = (float*)carve(256);

  // Layer 1
  gat_wprep<<<(NHEADS * HID * INDIM + 255) / 256, 256, 0, stream>>>(
      W1, WT1, INDIM, HID, NHEADS);
  gat_proj<INDIM, HID><<<dim3(NNODES / 16, NHEADS), 128, 0, stream>>>(
      X, WT1, h1f32, h1T, NNODES);
  gat_srcdst<<<(NHEADS * NNODES + 255) / 256, 256, 0, stream>>>(
      h1f32, a1, src1, dst1, NNODES, HID, NHEADS);
  gat_colmax<<<NHEADS, 256, 0, stream>>>(dst1, dmax1, NNODES);
  gat_attn1<<<NNODES / 16, 128, 0, stream>>>(adj, src1, dst1, dmax1, h1T, hcat);

  // Layer 2
  gat_wprep<<<(OUTD * NHEADS * HID + 255) / 256, 256, 0, stream>>>(
      W2, WT2, NHEADS * HID, OUTD, 1);
  gat_proj<NHEADS * HID, OUTD><<<dim3(NNODES / 16, OUTD / 64), 128, 0, stream>>>(
      hcat, WT2, h2f32, h2T, NNODES);
  gat_srcdst<<<(NNODES + 255) / 256, 256, 0, stream>>>(
      h2f32, a2, src2, dst2, NNODES, OUTD, 1);
  gat_colmax<<<1, 256, 0, stream>>>(dst2, dmax2, NNODES);
  gat_attn2<<<NNODES / 16, 128, 0, stream>>>(adj, src2, dst2, dmax2, h2T, out);
}
